// ConvEncLayer_22239340658704
// MI455X (gfx1250) — matrix operand order repaired and checked by reading, not compiled, not run on hardware
//
#include <hip/hip_runtime.h>
#include <hip/hip_bf16.h>

typedef __attribute__((ext_vector_type(16))) __bf16 v16bf;
typedef __attribute__((ext_vector_type(8)))  float  v8f;

#define HDIM 256
#define KNBR 8
#define TILE_M 16
#define LDS_STRIDE 264   // 256 + 8 bf16 pad: row stride 528B (16B aligned, bank-spread)

union BF16x16 { __bf16 b[16]; uint4 q[2]; v16bf v; };

// ---- weight f32->bf16 convert (k-contiguous per output column):  Wt[h][i] = (bf16)W[h][i] ----
__global__ void wtrans_kernel(const float* __restrict__ W, __bf16* __restrict__ Wt) {
    int tid = blockIdx.x * blockDim.x + threadIdx.x;
    if (tid < HDIM * HDIM) {
        int h = tid / HDIM;
        int i = tid % HDIM;
        Wt[h * HDIM + i] = (__bf16)W[h * HDIM + i];
    }
}

// load 16 consecutive bf16 (32 bytes, 16B-aligned) from LDS or global
__device__ inline v16bf load_frag16(const __bf16* p) {
    BF16x16 u;
    u.q[0] = *(const uint4*)(p);
    u.q[1] = *(const uint4*)(p + 8);
    return u.v;
}

__device__ inline void store_bf16x16(__bf16* dst, const float* v) {
    BF16x16 u;
#pragma unroll
    for (int j = 0; j < 16; ++j) u.b[j] = (__bf16)v[j];
    *(uint4*)(dst)     = u.q[0];
    *(uint4*)(dst + 8) = u.q[1];
}

// D[16 x 32] = rows[16 x 256](bf16, LDS) * W[256 x *wstride*] cols [ibase0, ibase0+32)
// A fragment: lanes 0-15 -> M=lane, K=hbase..+15 ; lanes 16-31 -> M=lane-16, K=hbase+16..31
// B fragment: lanes 0-15 -> N=lane, K=hbase..+15 ; lanes 16-31 -> N=lane-16, K=hbase+16..31
__device__ inline void gemm_tile(const __bf16* rows, const __bf16* W, int wstride,
                                 int lane, int ibase0, v8f& acc0, v8f& acc1) {
#pragma unroll
    for (int kk = 0; kk < 8; ++kk) {
        const int hbase = kk * 32;
        v16bf a  = load_frag16(rows + (lane & 15) * LDS_STRIDE + hbase + ((lane >> 4) << 4));
        v16bf b0 = load_frag16(W + (size_t)(ibase0 + (lane & 15)) * wstride + hbase + ((lane >> 4) << 4));
        v16bf b1 = load_frag16(W + (size_t)(ibase0 + 16 + (lane & 15)) * wstride + hbase + ((lane >> 4) << 4));
        acc0 = __builtin_amdgcn_wmma_f32_16x16x32_bf16(false, a, false, b0, (short)0, acc0, false, false);
        acc1 = __builtin_amdgcn_wmma_f32_16x16x32_bf16(false, a, false, b1, (short)0, acc1, false, false);
    }
}

__device__ inline float fast_sigmoid(float x) {
    return 1.0f / (1.0f + __expf(-x));
}

__global__ __launch_bounds__(256)
void jtconv_kernel(const float* __restrict__ prev,
                   const __bf16* __restrict__ WtU, const __bf16* __restrict__ WtV,
                   const __bf16* __restrict__ WtA, const __bf16* __restrict__ WtB,
                   const int* __restrict__ parent_idxs,
                   const int* __restrict__ child_idxs,
                   const unsigned char* __restrict__ child_mask,
                   float* __restrict__ out, int N) {
    __shared__ __bf16 wb_lds[HDIM * LDS_STRIDE];     // W_B^T staged once, used 8x (132KB)
    __shared__ __bf16 ph_lds[TILE_M * LDS_STRIDE];   // parent tile
    __shared__ __bf16 ch_lds[TILE_M * LDS_STRIDE];   // per-slot child tile; reused for child-sum
    __shared__ int           cidx_lds[TILE_M * KNBR];
    __shared__ unsigned char cmask_lds[TILE_M * KNBR];
    __shared__ int           prow_lds[TILE_M];

    const int tid  = threadIdx.x;
    const int lane = tid & 31;
    const int wave = tid >> 5;
    const int r    = tid >> 4;        // tile row this thread stages (0..15)
    const int col0 = (tid & 15) * 16; // 16-float chunk this thread stages
    const int rowbase = blockIdx.x * TILE_M;

    // ---- async-stage W_B^T into LDS: 8192 x 16B chunks, ASYNCcnt-tracked DMA ----
    // chunk c = tid + it*256: global addr advances by 4096B/iter, LDS by 8 padded rows.
    {
        unsigned loff = (unsigned)(size_t)&wb_lds[(tid >> 5) * LDS_STRIDE + (tid & 31) * 8];
        unsigned long long ga = (unsigned long long)(size_t)WtB + (size_t)tid * 16;
#pragma unroll
        for (int it = 0; it < (HDIM * HDIM * 2) / (16 * 256); ++it) {
            asm volatile("global_load_async_to_lds_b128 %0, %1, off"
                         :: "v"(loff), "v"(ga) : "memory");
            loff += 8u * (LDS_STRIDE * 2);
            ga   += 4096ull;
        }
    }

    // ---- stage parent rows (gather via parent_idxs) as bf16 into LDS ----
    {
        int grow = rowbase + r;
        if (grow >= N) grow = N - 1;
        int prow = parent_idxs[grow];
        if ((tid & 15) == 0) prow_lds[r] = prow;
        const float4* s = (const float4*)(prev + (size_t)prow * HDIM + col0);
        float v[16];
#pragma unroll
        for (int q = 0; q < 4; ++q) {
            float4 f = s[q];
            v[4 * q + 0] = f.x; v[4 * q + 1] = f.y; v[4 * q + 2] = f.z; v[4 * q + 3] = f.w;
        }
        store_bf16x16(&ph_lds[r * LDS_STRIDE + col0], v);
    }
    if (tid < TILE_M * KNBR) {
        int grow = rowbase + (tid / KNBR);
        if (grow >= N) grow = N - 1;
        int slot = tid % KNBR;
        cidx_lds[tid]  = child_idxs[(size_t)grow * KNBR + slot];
        cmask_lds[tid] = child_mask[(size_t)grow * KNBR + slot];
    }
    asm volatile("s_wait_asynccnt 0" ::: "memory");  // our async chunks landed
    __syncthreads();

    const int ibase0 = wave * 32;   // this wave owns output columns [ibase0, ibase0+32)

    // ---- A projection only (needed inside the sigmoid loop); U/V deferred to
    //      shrink live ranges: peak register state in the k-loop stays small ----
    v8f accA0 = {}, accA1 = {};
    gemm_tile(ph_lds, WtA, HDIM, lane, ibase0, accA0, accA1);

    v8f eta0 = {}, eta1 = {};
    float csum[16];
#pragma unroll
    for (int j = 0; j < 16; ++j) csum[j] = 0.0f;

    // ---- per child slot: gather -> B projection (weights from LDS) -> sigmoid ----
#pragma unroll 1
    for (int k = 0; k < KNBR; ++k) {
        __syncthreads();  // previous WMMA reads of ch_lds complete before overwrite
        {
            int   ci = cidx_lds[r * KNBR + k];
            bool  m  = cmask_lds[r * KNBR + k] != 0;
            float v[16];
            if (m) {
                const float4* s = (const float4*)(prev + (size_t)ci * HDIM + col0);
#pragma unroll
                for (int q = 0; q < 4; ++q) {
                    float4 f = s[q];
                    v[4 * q + 0] = f.x; v[4 * q + 1] = f.y; v[4 * q + 2] = f.z; v[4 * q + 3] = f.w;
                }
            } else {
#pragma unroll
                for (int j = 0; j < 16; ++j) v[j] = 0.0f;
            }
#pragma unroll
            for (int j = 0; j < 16; ++j) csum[j] += v[j];
            store_bf16x16(&ch_lds[r * LDS_STRIDE + col0], v);
        }
        __syncthreads();

        v8f b0 = {}, b1 = {};
        gemm_tile(ch_lds, wb_lds, LDS_STRIDE, lane, ibase0, b0, b1);
#pragma unroll
        for (int j = 0; j < 8; ++j) {
            eta0[j] += fast_sigmoid(accA0[j] + b0[j]);
            eta1[j] += fast_sigmoid(accA1[j] + b1[j]);
        }
    }

    // ---- V projection of the child-sum (linear: Σ_k V(ch_k) == V(Σ_k ch_k)) ----
    __syncthreads();
    store_bf16x16(&ch_lds[r * LDS_STRIDE + col0], csum);
    __syncthreads();

    v8f accV0 = {}, accV1 = {};
    gemm_tile(ch_lds, WtV, HDIM, lane, ibase0, accV0, accV1);

    // ---- U projection last (short live range) ----
    v8f accU0 = {}, accU1 = {};
    gemm_tile(ph_lds, WtU, HDIM, lane, ibase0, accU0, accU1);

    // ---- combine + ReLU + scatter ----
    // D layout: VGPR j, lanes 0-15 -> M=j, N=lane; lanes 16-31 -> M=j+8, N=lane-16
    const int ncol = lane & 15;
    const int mhi  = (lane >> 4) * 8;
#pragma unroll
    for (int j = 0; j < 8; ++j) {
        int mrow = j + mhi;
        int grow = rowbase + mrow;
        if (grow < N) {
            float* op = out + (size_t)prow_lds[mrow] * HDIM;
            float o0 = accU0[j] + eta0[j] + accV0[j];
            float o1 = accU1[j] + eta1[j] + accV1[j];
            op[ibase0 + ncol]      = o0 > 0.0f ? o0 : 0.0f;
            op[ibase0 + 16 + ncol] = o1 > 0.0f ? o1 : 0.0f;
        }
    }
}

extern "C" void kernel_launch(void* const* d_in, const int* in_sizes, int n_in,
                              void* d_out, int out_size, void* d_ws, size_t ws_size,
                              hipStream_t stream) {
    const float* prev        = (const float*)d_in[0];
    const float* W_U         = (const float*)d_in[1];
    const float* W_V         = (const float*)d_in[2];
    const float* W_A         = (const float*)d_in[3];
    const float* W_B         = (const float*)d_in[4];
    const int*   parent_idxs = (const int*)d_in[5];
    const int*   child_idxs  = (const int*)d_in[6];
    const unsigned char* child_mask = (const unsigned char*)d_in[7]; // jnp.bool_ = 1 byte

    const int N = in_sizes[0] / HDIM;

    // workspace: 4 transposed bf16 weights, 128KB each (total 512KB)
    char* ws = (char*)d_ws;
    __bf16* WtU = (__bf16*)(ws + 0 * (size_t)HDIM * HDIM * 2);
    __bf16* WtV = (__bf16*)(ws + 1 * (size_t)HDIM * HDIM * 2);
    __bf16* WtA = (__bf16*)(ws + 2 * (size_t)HDIM * HDIM * 2);
    __bf16* WtB = (__bf16*)(ws + 3 * (size_t)HDIM * HDIM * 2);

    const int wgrid = (HDIM * HDIM + 255) / 256;
    wtrans_kernel<<<wgrid, 256, 0, stream>>>(W_U, WtU);
    wtrans_kernel<<<wgrid, 256, 0, stream>>>(W_V, WtV);
    wtrans_kernel<<<wgrid, 256, 0, stream>>>(W_A, WtA);
    wtrans_kernel<<<wgrid, 256, 0, stream>>>(W_B, WtB);

    // rows not touched by the scatter must read as zero
    hipMemsetAsync(d_out, 0, (size_t)out_size * sizeof(float), stream);

    const int grid = (N + TILE_M - 1) / TILE_M;
    jtconv_kernel<<<grid, 256, 0, stream>>>(prev, WtU, WtV, WtA, WtB,
                                            parent_idxs, child_idxs, child_mask,
                                            (float*)d_out, N);
}
